// TrackPairingPredictionNetwork_88768384074460
// MI455X (gfx1250) — matrix operand order repaired and checked by reading, not compiled, not run on hardware
//
#include <hip/hip_runtime.h>

typedef __attribute__((ext_vector_type(16))) _Float16 v16h;
typedef __attribute__((ext_vector_type(8)))  float    v8f;

// Problem constants: J=128 jets, T=64 tracks, D=64 embed; pairs = 128*64*64.
#define NJ 128
#define NT 64
#define ND 64
#define NROWS (NJ * NT)          // 8192 (j,i) rows

// A-operand K pattern for 16-bit 16x32 WMMA A matrix (ISA 7.12.2):
// lanes 0-15 hold K = {0..7, 16..23}, lanes 16-31 hold K = {8..15, 24..31}.
#define FPAT(e, hs) (((e) < 8) ? ((hs) * 8 + (e)) : (16 + (hs) * 8 + (e) - 8))

// Row permutation for W2^T tiles: A-row r of tile mt holds hidden unit
// P(mt,r), chosen so hs=0 lanes end up holding units 0..15 and hs=1 lanes
// units 16..31 -> layer-3 B-operand builds from the lane's OWN registers.
#define PERM(mt, r) (((r) < 8) ? ((mt) * 8 + (r)) : (16 + (mt) * 8 + (r) - 8))

// ds_swizzle immediate: group-of-32 mode, xor_mask=0x10, and_mask=0x1f
// (swap 16-lane halves).
#define SWAPX16_IMM 0x401f

// single-instruction relu (v_med3_f32), avoids the canonicalize+max pair
__device__ __forceinline__ float relu1(float x) {
  return __builtin_amdgcn_fmed3f(x, 0.f, __builtin_inff());
}

// ---------------------------------------------------------------------------
// Stage 1: factored layer-1 precompute (f32 FMAs, trivially cheap):
//   Aout[j*64+i][f] = trk[j,i]@W1[0:64]   + jet[j]@W1[128:192] + b1   (f16)
//   Bout[j*64+k][f] = trk[j,k]@W1[64:128]                            (f16)
// ---------------------------------------------------------------------------
__global__ __launch_bounds__(256) void tpn_stage1(
    const float* __restrict__ jet, const float* __restrict__ trk,
    const float* __restrict__ W1, const float* __restrict__ b1,
    _Float16* __restrict__ Aout, _Float16* __restrict__ Bout)
{
  const int gid = blockIdx.x * blockDim.x + threadIdx.x;  // 0 .. 8192*64-1
  const int row = gid >> 6;       // (j,i) row
  const int f   = gid & 63;       // hidden feature
  const int jj  = row >> 6;       // jet index

  const float* t  = trk + row * ND;
  const float* je = jet + jj * ND;

  float a = b1[f];                // fold b1 into A
  float b = 0.f;
  #pragma unroll 8
  for (int d = 0; d < ND; ++d) {
    const float td = t[d];
    a = fmaf(td,    W1[d * 64 + f],          a);   // W1a column f (coalesced in f)
    b = fmaf(td,    W1[(64 + d) * 64 + f],   b);   // W1b
    a = fmaf(je[d], W1[(128 + d) * 64 + f],  a);   // W1c (jet term folded into A)
  }
  Aout[gid] = (_Float16)a;
  Bout[gid] = (_Float16)b;
}

// ---------------------------------------------------------------------------
// Stage 2: fused layers 1/2/3/4 + sigmoid, computed TRANSPOSED with
// permuted W2^T rows. One wave handles 2 consecutive (j,i) rows x 4 k-tiles
// of 16 pairs. Layers 2/3 = v_wmma_f32_16x16x32_f16; all glue is
// register-only packed-f16 / med3 except one immediate ds_swizzle per tile.
// ---------------------------------------------------------------------------
__global__ __launch_bounds__(256) void tpn_stage2(
    const _Float16* __restrict__ A, const _Float16* __restrict__ B,
    const float* __restrict__ W2, const float* __restrict__ b2,
    const float* __restrict__ W3, const float* __restrict__ b3,
    const float* __restrict__ W4, const float* __restrict__ b4,
    float* __restrict__ out)
{
  const int lane = threadIdx.x & 31;
  const int wave = threadIdx.x >> 5;
  const int m    = lane & 15;     // pair column (B/D layouts: col = lane%16)
  const int hs   = lane >> 4;     // lane half

  const int t2 = blockIdx.x * 8 + wave;  // 0..4095
  const int j  = t2 >> 5;                // jet
  const int ib = (t2 & 31) * 2;          // first of 2 i rows for this wave

  // ---- W2^T as WMMA A-operands, rows permuted by PERM
  v16h w2a[2][2];
  #pragma unroll
  for (int mt = 0; mt < 2; ++mt)
    #pragma unroll
    for (int kp = 0; kp < 2; ++kp)
      #pragma unroll
      for (int e = 0; e < 16; ++e)
        w2a[mt][kp][e] = (_Float16)W2[(kp * 32 + hs * 16 + e) * 32 + PERM(mt, m)];

  // ---- W3^T as WMMA A-operand (unpermuted): row M = m, K = 16*hs + e
  v16h w3a;
  #pragma unroll
  for (int e = 0; e < 16; ++e)
    w3a[e] = (_Float16)W3[(hs * 16 + e) * 16 + m];

  // ---- bias seeds as v8f WMMA C operands (rows match permuted D layout)
  v8f c2s[2], c3s;
  float w4r[8];
  #pragma unroll
  for (int mt = 0; mt < 2; ++mt)
    #pragma unroll
    for (int v = 0; v < 8; ++v)
      c2s[mt][v] = b2[16 * hs + mt * 8 + v];   // b2[PERM(mt, v+8hs)]
  #pragma unroll
  for (int v = 0; v < 8; ++v) {
    c3s[v] = b3[8 * hs + v];
    w4r[v] = W4[8 * hs + v];
  }
  const float b4v = b4[0];

  // ---- per-i invariant layer-1 partials (contiguous 16-half runs)
  v16h basev[2][2];
  #pragma unroll
  for (int ii = 0; ii < 2; ++ii) {
    const _Float16* Arow = A + (j * 64 + ib + ii) * 64;
    basev[ii][0] = *(const v16h*)(Arow + hs * 16);
    basev[ii][1] = *(const v16h*)(Arow + 32 + hs * 16);
  }
  const v16h vzero = {};

  for (int kt = 0; kt < 4; ++kt) {
    // B row depends only on (j, k): load once, reuse for both i rows
    const _Float16* Brow = B + (j * 64 + kt * 16 + m) * 64;
    v16h browv[2];
    browv[0] = *(const v16h*)(Brow + hs * 16);
    browv[1] = *(const v16h*)(Brow + 32 + hs * 16);

    #pragma unroll
    for (int ii = 0; ii < 2; ++ii) {
      // layer 1: h1^T B-operand entirely in packed f16
      v16h b2op[2];
      b2op[0] = __builtin_elementwise_max(basev[ii][0] + browv[0], vzero);
      b2op[1] = __builtin_elementwise_max(basev[ii][1] + browv[1], vzero);

      // layer 2: h2^T [32x16] = permuted W2^T @ h1^T, seeded with permuted b2
      v8f d2[2];
      #pragma unroll
      for (int mt = 0; mt < 2; ++mt) {
        d2[mt] = __builtin_amdgcn_wmma_f32_16x16x32_f16(false, w2a[mt][0], false, b2op[0], (short)0, c2s[mt], false, false);
        d2[mt] = __builtin_amdgcn_wmma_f32_16x16x32_f16(false, w2a[mt][1], false, b2op[1], (short)0, d2[mt],  false, false);
      }

      // layer-3 B-operand from own registers: convert (v_cvt_pk_f16_f32)
      // THEN relu in packed f16 (cvt output is canonical -> no canonicalize
      // pair; relu and cvt commute exactly).
      v16h b3raw;
      #pragma unroll
      for (int e = 0; e < 8; ++e) {
        b3raw[e]     = (_Float16)d2[0][e];   // K = 16hs + e
        b3raw[e + 8] = (_Float16)d2[1][e];   // K = 16hs + 8 + e
      }
      const v16h b3op = __builtin_elementwise_max(b3raw, vzero);

      // layer 3: h3^T [16x16] = W3^T @ h2^T, seeded with b3
      v8f d3 = __builtin_amdgcn_wmma_f32_16x16x32_f16(false, w3a, false, b3op, (short)0, c3s, false, false);

      // layer 4 (N=1): med3-relu + 8 register FMAs, one half-swap swizzle
      float x = 0.f;
      #pragma unroll
      for (int v = 0; v < 8; ++v)
        x = fmaf(relu1(d3[v]), w4r[v], x);
      x += __int_as_float(__builtin_amdgcn_ds_swizzle(__float_as_int(x), SWAPX16_IMM));

      if (hs == 0) {
        const float lg = x + b4v;
        out[(j * 64 + ib + ii) * 64 + kt * 16 + m] = 1.f / (1.f + __expf(-lg));
      }
    }
  }
}

// ---------------------------------------------------------------------------
extern "C" void kernel_launch(void* const* d_in, const int* in_sizes, int n_in,
                              void* d_out, int out_size, void* d_ws, size_t ws_size,
                              hipStream_t stream)
{
  const float* jet = (const float*)d_in[0];
  const float* trk = (const float*)d_in[1];
  const float* W1  = (const float*)d_in[2];
  const float* b1  = (const float*)d_in[3];
  const float* W2  = (const float*)d_in[4];
  const float* b2  = (const float*)d_in[5];
  const float* W3  = (const float*)d_in[6];
  const float* b3  = (const float*)d_in[7];
  const float* W4  = (const float*)d_in[8];
  const float* b4  = (const float*)d_in[9];

  char* ws = (char*)d_ws;
  _Float16* Abuf = (_Float16*)ws;                              // 8192*64 f16 = 1 MB
  _Float16* Bbuf = (_Float16*)(ws + (size_t)NROWS * 64 * 2);   // 1 MB  (total 2 MB)

  tpn_stage1<<<(NROWS * 64) / 256, 256, 0, stream>>>(jet, trk, W1, b1, Abuf, Bbuf);
  tpn_stage2<<<NROWS / 16, 256, 0, stream>>>(Abuf, Bbuf, W2, b2, W3, b3, W4, b4,
                                             (float*)d_out);
}
